// GlobalPointer_652835029496
// MI455X (gfx1250) — matrix operand order repaired and checked by reading, not compiled, not run on hardware
//
#include <hip/hip_runtime.h>

// ---------------------------------------------------------------------------
// Reference shapes: B=8, L=1024, D=768, HEADS=12, HEAD_SIZE=64, N = 1536
// out = causal-masked scaled QK^T logits, (8,12,1024,1024) fp32 (402 MB).
// Output-BW bound (~17us floor @23.3TB/s) -> fp16 WMMA everywhere, one pass
// over the output with fused epilogue. Kernel 2 stages Q/K tiles with the
// gfx1250 async global->LDS path (ASYNCcnt) via the clang builtin.
// ---------------------------------------------------------------------------

typedef __attribute__((ext_vector_type(16))) _Float16 v16h;
typedef __attribute__((ext_vector_type(8)))  _Float16 h8;
typedef __attribute__((ext_vector_type(8)))  float    v8f;

#define NEG_INF 1.0e12f
#define LOG2_10000 13.287712379549449f

#if __has_builtin(__builtin_amdgcn_global_load_async_to_lds_b128) && \
    __has_builtin(__builtin_amdgcn_s_wait_asynccnt)
#define HAVE_ASYNC_LDS 1
#else
#define HAVE_ASYNC_LDS 0
#endif

#if HAVE_ASYNC_LDS
// Builtin signature (from hipcc diagnostic): param 1 = addrspace(1) pointer to
// 16-byte int vector (global src), param 2 = addrspace(3) pointer (LDS dst),
// then imm offset and imm cpol.
typedef __attribute__((__vector_size__(4 * sizeof(int)))) int b128_t;
#define ASYNC_CPY_B128(dst_lds, src_glob)                                      \
    __builtin_amdgcn_global_load_async_to_lds_b128(                            \
        (__attribute__((address_space(1))) b128_t*)(src_glob),                 \
        (__attribute__((address_space(3))) b128_t*)(dst_lds), 0, 0)
#endif

union V16H { v16h v; h8 h[2]; };

// A-fragment (16x32 f16, MxK), row-major LDS tile with stride lda (halves).
// ISA layout: lanes 0-15: M=lane, K in {0..7,16..23}; lanes 16-31: K in {8..15,24..31}.
__device__ __forceinline__ v16h lds_a_frag(const _Float16* base, int lda, int lane) {
    const int m  = lane & 15;
    const int kh = (lane & 16) ? 8 : 0;
    V16H u;
    u.h[0] = *(const h8*)(base + m * lda + kh);
    u.h[1] = *(const h8*)(base + m * lda + kh + 16);
    return u.v;
}

// B-fragment (32x16 f16, KxN), LDS tile stored [n][k] with stride ldb (halves).
// ISA layout: lanes 0-15: N=lane, K in {0..7,16..23}; lanes 16-31: N=lane-16, K in {8..15,24..31}.
__device__ __forceinline__ v16h lds_b_frag(const _Float16* base, int ldb, int lane) {
    const int n  = lane & 15;
    const int kh = (lane & 16) ? 8 : 0;
    V16H u;
    u.h[0] = *(const h8*)(base + n * ldb + kh);
    u.h[1] = *(const h8*)(base + n * ldb + kh + 16);
    return u.v;
}

// ---------------------------------------------------------------------------
// Kernel 1: proj = x @ W + b  (8192x768 . 768x1536), fused interleaved RoPE,
// writes Q and K as fp16 in [b*H + h][l][s] layout.
// Block: 256 thr (8 waves). Tile BM=64 x BN=128 (one full head) x BK=32.
// Wave grid 4(M) x 2(N): each wave 16x64 = 4 WMMA accumulators.
// ---------------------------------------------------------------------------
__global__ __launch_bounds__(256)
void proj_rope_kernel(const float* __restrict__ x, const float* __restrict__ W,
                      const float* __restrict__ bias,
                      _Float16* __restrict__ qbuf, _Float16* __restrict__ kbuf) {
    __shared__ __align__(16) _Float16 As[64][40];    // [m][k], pad to 40 halves
    __shared__ __align__(16) _Float16 Bs[128][40];   // [n][k] (W transposed)

    const int t    = threadIdx.x;
    const int lane = t & 31;
    const int wid  = t >> 5;
    const int m0   = blockIdx.y * 64;    // row in flattened (B*L, D)
    const int n0   = blockIdx.x * 128;   // head-aligned column block

    v8f acc[4] = {};

    for (int kt = 0; kt < 768; kt += 32) {
        // ---- stage A tile 64x32 (fp32 -> fp16), coalesced float4 loads
        {
            const int row = t >> 2;
            const int cs  = (t & 3) * 8;
            const float* src = x + (size_t)(m0 + row) * 768 + kt + cs;
            float4 f0 = *(const float4*)(src);
            float4 f1 = *(const float4*)(src + 4);
            _Float16* d = &As[row][cs];
            d[0] = (_Float16)f0.x; d[1] = (_Float16)f0.y;
            d[2] = (_Float16)f0.z; d[3] = (_Float16)f0.w;
            d[4] = (_Float16)f1.x; d[5] = (_Float16)f1.y;
            d[6] = (_Float16)f1.z; d[7] = (_Float16)f1.w;
        }
        // ---- stage B tile 32x128, transposed into Bs[n][k]
#pragma unroll
        for (int it = 0; it < 4; ++it) {
            const int kr = (t >> 5) + it * 8;
            const int nc = (t & 31) * 4;
            float4 w = *(const float4*)(W + (size_t)(kt + kr) * 1536 + n0 + nc);
            Bs[nc + 0][kr] = (_Float16)w.x;
            Bs[nc + 1][kr] = (_Float16)w.y;
            Bs[nc + 2][kr] = (_Float16)w.z;
            Bs[nc + 3][kr] = (_Float16)w.w;
        }
        if (kt + 32 < 768) {  // hint next A k-slice toward the caches
            __builtin_prefetch(x + (size_t)(m0 + (t >> 2)) * 768 + kt + 32, 0, 3);
        }
        __syncthreads();

        const v16h a = lds_a_frag(&As[(wid & 3) * 16][0], 40, lane);
#pragma unroll
        for (int j = 0; j < 4; ++j) {
            const v16h b = lds_b_frag(&Bs[(wid >> 2) * 64 + j * 16][0], 40, lane);
            acc[j] = __builtin_amdgcn_wmma_f32_16x16x32_f16(
                false, a, false, b, (short)0, acc[j], false, false);
        }
        __syncthreads();
    }

    // ---- epilogue: + bias, interleaved RoPE (pair swap via shfl_xor 1),
    //      fp16 store to Q or K in [bh][l][s] layout.
    const int mw = m0 + (wid & 3) * 16;
#pragma unroll
    for (int j = 0; j < 4; ++j) {
        const int gn = n0 + (wid >> 2) * 64 + j * 16 + (lane & 15);
        const int h  = gn >> 7;          // head index
        const int c  = gn & 127;         // column within head block
        const int qk = (c >> 6) & 1;     // 0 = Q half, 1 = K half
        const int s  = c & 63;           // position within head_size
        const float inv = exp2f(-(float)(s & ~1) * (LOG2_10000 / 64.0f));
        const float bv  = bias[gn];
        _Float16* dstbuf = qk ? kbuf : qbuf;
#pragma unroll
        for (int v = 0; v < 8; ++v) {
            const int gm = mw + v + ((lane >> 4) << 3);
            const int bi = gm >> 10;
            const int l  = gm & 1023;
            float sn, cs;
            sincosf((float)l * inv, &sn, &cs);
            const float val = acc[j][v] + bv;
            const float par = __shfl_xor(val, 1);   // partner of interleaved pair
            const float ro  = (s & 1) ? (val * cs + par * sn)
                                      : (val * cs - par * sn);
            dstbuf[((size_t)(bi * 12 + h) * 1024 + l) * 64 + s] = (_Float16)ro;
        }
    }
}

// ---------------------------------------------------------------------------
// Kernel 2: logits[b,h,m,n] = sum_s Q[bh,m,s]*K[bh,n,s], fused mask/causal/scale.
// Block: 256 thr, tile 128x128 per (b,h). K=64 -> two WMMA k-steps.
// Wave grid 4(M) x 2(N): each wave 32x64 = 2x4 WMMA accumulators.
// Q/K tiles staged via async global->LDS (ASYNCcnt) when available.
// ---------------------------------------------------------------------------
__global__ __launch_bounds__(256)
void attn_logits_kernel(const _Float16* __restrict__ qbuf,
                        const _Float16* __restrict__ kbuf,
                        const int* __restrict__ mask,
                        float* __restrict__ out) {
    __shared__ __align__(16) _Float16 Qs[128][72];
    __shared__ __align__(16) _Float16 Ks[128][72];

    const int t    = threadIdx.x;
    const int lane = t & 31;
    const int wid  = t >> 5;
    const int bh   = blockIdx.z;
    const int bi   = bh / 12;
    const int m0   = blockIdx.y * 128;
    const int n0   = blockIdx.x * 128;

    const _Float16* qg = qbuf + (size_t)bh * 1024 * 64;
    const _Float16* kg = kbuf + (size_t)bh * 1024 * 64;

    {   // stage Q and K tiles (128 x 64 halves each)
        const int r  = t >> 2;
        const int sg = (t & 3) * 16;
#if HAVE_ASYNC_LDS
#pragma unroll
        for (int rr = 0; rr < 128; rr += 64) {
            const _Float16* qs = qg + (size_t)(m0 + r + rr) * 64 + sg;
            ASYNC_CPY_B128(&Qs[r + rr][sg + 0], qs + 0);
            ASYNC_CPY_B128(&Qs[r + rr][sg + 8], qs + 8);
            const _Float16* ks = kg + (size_t)(n0 + r + rr) * 64 + sg;
            ASYNC_CPY_B128(&Ks[r + rr][sg + 0], ks + 0);
            ASYNC_CPY_B128(&Ks[r + rr][sg + 8], ks + 8);
        }
        __builtin_amdgcn_s_wait_asynccnt(0);
#else
#pragma unroll
        for (int rr = 0; rr < 128; rr += 64) {
            const h8* qs = (const h8*)(qg + (size_t)(m0 + r + rr) * 64 + sg);
            *(h8*)&Qs[r + rr][sg + 0] = qs[0];
            *(h8*)&Qs[r + rr][sg + 8] = qs[1];
            const h8* ks = (const h8*)(kg + (size_t)(n0 + r + rr) * 64 + sg);
            *(h8*)&Ks[r + rr][sg + 0] = ks[0];
            *(h8*)&Ks[r + rr][sg + 8] = ks[1];
        }
#endif
    }
    __syncthreads();

    v8f acc[2][4] = {};
    const int wm = (wid & 3) * 32;
    const int wn = (wid >> 2) * 64;
#pragma unroll
    for (int kb = 0; kb < 64; kb += 32) {
        const v16h a0 = lds_a_frag(&Qs[wm][kb],      72, lane);
        const v16h a1 = lds_a_frag(&Qs[wm + 16][kb], 72, lane);
#pragma unroll
        for (int nt = 0; nt < 4; ++nt) {
            const v16h b = lds_b_frag(&Ks[wn + nt * 16][kb], 72, lane);
            acc[0][nt] = __builtin_amdgcn_wmma_f32_16x16x32_f16(
                false, a0, false, b, (short)0, acc[0][nt], false, false);
            acc[1][nt] = __builtin_amdgcn_wmma_f32_16x16x32_f16(
                false, a1, false, b, (short)0, acc[1][nt], false, false);
        }
    }

    // ---- epilogue: row/col mask, causal (keep m<=n), scale 1/sqrt(64)
#pragma unroll
    for (int mt = 0; mt < 2; ++mt) {
#pragma unroll
        for (int nt = 0; nt < 4; ++nt) {
            const int gn   = n0 + wn + nt * 16 + (lane & 15);
            const float mc = (float)mask[bi * 1024 + gn];
#pragma unroll
            for (int v = 0; v < 8; ++v) {
                const int gm   = m0 + wm + mt * 16 + v + ((lane >> 4) << 3);
                const float mr = (float)mask[bi * 1024 + gm];
                float val = acc[mt][nt][v];
                val = val * mr - NEG_INF * (1.0f - mr);
                val = val * mc - NEG_INF * (1.0f - mc);
                if (gm > gn) val -= NEG_INF;
                val *= 0.125f;
                out[(size_t)bh * 1024 * 1024 + (size_t)gm * 1024 + gn] = val;
            }
        }
    }
}

// ---------------------------------------------------------------------------
extern "C" void kernel_launch(void* const* d_in, const int* in_sizes, int n_in,
                              void* d_out, int out_size, void* d_ws, size_t ws_size,
                              hipStream_t stream) {
    (void)in_sizes; (void)n_in; (void)out_size; (void)ws_size;

    const float* x    = (const float*)d_in[0];   // (8,1024,768) f32
    const float* W    = (const float*)d_in[1];   // (768,1536)   f32
    const float* bias = (const float*)d_in[2];   // (1536,)      f32
    const int*   mask = (const int*)d_in[3];     // (8,1024)     i32
    float* out = (float*)d_out;                  // (8,12,1024,1024) f32

    _Float16* qbuf = (_Float16*)d_ws;                         // 12.6 MB fp16
    _Float16* kbuf = qbuf + (size_t)8 * 12 * 1024 * 64;       // 12.6 MB fp16

    dim3 g1(1536 / 128, 8192 / 64);     // (12, 128)
    proj_rope_kernel<<<g1, 256, 0, stream>>>(x, W, bias, qbuf, kbuf);

    dim3 g2(1024 / 128, 1024 / 128, 8 * 12);   // (8, 8, 96)
    attn_logits_kernel<<<g2, 256, 0, stream>>>(qbuf, kbuf, mask, out);
}
